// NisuyNN_30081950941733
// MI455X (gfx1250) — matrix operand order repaired and checked by reading, not compiled, not run on hardware
//
#include <hip/hip_runtime.h>
#include <hip/hip_bf16.h>

#define DIM   128
#define NN    16
#define BB    8
#define HH    4096
#define PP    256           // NN*NN
#define MM    2048          // PP*BB
#define SLOPE 0.01f
#define PI_ITERS 50

typedef __attribute__((ext_vector_type(16))) __bf16 v16bf;
typedef __attribute__((ext_vector_type(8)))  __bf16 v8bf;
typedef __attribute__((ext_vector_type(8)))  float  v8f;

static __device__ __forceinline__ v16bf cat16(v8bf lo, v8bf hi) {
    return __builtin_shufflevector(lo, hi, 0, 1, 2, 3, 4, 5, 6, 7,
                                           8, 9, 10, 11, 12, 13, 14, 15);
}

// ---------------------------------------------------------------------------
// Build X (2048 x 256) bf16, replicating the reference's stack+reshape layout:
// row r = p*8 + b':
//   b' in [0,4):  X[r] = concat(nodes[2b'][s], nodes[2b'+1][s]),  s = p>>4
//   b' in [4,8):  X[r] = concat(nodes[2(b'-4)][t], nodes[2(b'-4)+1][t]), t = p&15
// ---------------------------------------------------------------------------
__global__ void build_x_kernel(const float* __restrict__ nodes, __bf16* __restrict__ Xb) {
    int idx = blockIdx.x * 256 + threadIdx.x;   // 2048*256 total
    int e = idx & 255;
    int r = idx >> 8;
    int bp = r & 7;          // b'
    int p  = r >> 3;
    int d  = e & 127;
    int hi = e >> 7;         // which half of the 256-wide feature
    int node, batch;
    if (bp < 4) { node = p >> 4;  batch = 2 * bp + hi; }
    else        { node = p & 15;  batch = 2 * (bp - 4) + hi; }
    Xb[idx] = (__bf16)nodes[((size_t)batch * NN + node) * DIM + d];
}

// ---------------------------------------------------------------------------
// Transpose + fp32->bf16: Wt[n*K + k] = (bf16) W[k*N + n].  LDS 32x32 tiles.
// ---------------------------------------------------------------------------
__global__ void transpose_cvt_kernel(const float* __restrict__ W, __bf16* __restrict__ Wt,
                                     int K, int N) {
    __shared__ float tile[32][33];
    int tx = threadIdx.x & 31;
    int ty = threadIdx.x >> 5;          // 0..7
    int k0 = blockIdx.x * 32;
    int n0 = blockIdx.y * 32;
#pragma unroll
    for (int r = 0; r < 4; ++r)
        tile[ty + r * 8][tx] = W[(size_t)(k0 + ty + r * 8) * N + (n0 + tx)];
    __syncthreads();
#pragma unroll
    for (int r = 0; r < 4; ++r)
        Wt[(size_t)(n0 + ty + r * 8) * K + (k0 + tx)] = (__bf16)tile[tx][ty + r * 8];
}

// ---------------------------------------------------------------------------
// GEMM fragments: one K-step (32) worth of 2 A-frags + 4 B-frags.
// ---------------------------------------------------------------------------
struct FragSet {
    v8bf a0l, a0h, a1l, a1h;
    v8bf bl[4], bh[4];
};

template <int K>
static __device__ __forceinline__ void load_set(FragSet& f, const __bf16* __restrict__ aP,
                                                const __bf16* __restrict__ bP, int k0) {
    // A per ISA 16-bit A 16x32 layout; second A tile at constant offset 16*K.
    f.a0l = *(const v8bf*)(aP + k0);
    f.a0h = *(const v8bf*)(aP + k0 + 16);
    f.a1l = *(const v8bf*)(aP + k0 + 16 * K);
    f.a1h = *(const v8bf*)(aP + k0 + 16 * K + 16);
    // B per ISA 16-bit B 32x16 layout; column group j at constant offset j*16*K.
#pragma unroll
    for (int j = 0; j < 4; ++j) {
        f.bl[j] = *(const v8bf*)(bP + k0 + j * 16 * K);
        f.bh[j] = *(const v8bf*)(bP + k0 + j * 16 * K + 16);
    }
}

static __device__ __forceinline__ void compute_set(const FragSet& f, v8f acc[2][4]) {
    v16bf A0 = cat16(f.a0l, f.a0h);
    v16bf A1 = cat16(f.a1l, f.a1h);
#pragma unroll
    for (int j = 0; j < 4; ++j) {
        v16bf Bj = cat16(f.bl[j], f.bh[j]);
        acc[0][j] = __builtin_amdgcn_wmma_f32_16x16x32_bf16(
            false, A0, false, Bj, (short)0, acc[0][j], false, false);
        acc[1][j] = __builtin_amdgcn_wmma_f32_16x16x32_bf16(
            false, A1, false, Bj, (short)0, acc[1][j], false, false);
    }
}

// ---------------------------------------------------------------------------
// GEMM: out = act(X(MxK,bf16 rowmajor) * W(KxN) + bias), W given as Wt (NxK).
// Wave computes a 32x64 tile (8 WMMAs / K-step); 8 waves/block arranged
// 4(M) x 2(N) -> 128x128 block tile.  Two fragment sets are software-pipelined
// in a body that is already manually 2x-unrolled (buffer swap by renaming);
// the loop itself is pinned rolled (#pragma unroll 1) so exactly two FragSets
// stay live -> no spills.  K is a compile-time constant so A/B strides are
// VMEM immediates -> 2 live base pointers.  K must be a multiple of 64.
// LAST=false: leaky-ReLU -> bf16 out.  LAST=true: sigmoid(leaky-ReLU) -> f32.
// ---------------------------------------------------------------------------
template <int K, bool LAST>
__global__ __launch_bounds__(256)
void gemm_wmma_kernel(const __bf16* __restrict__ X, const __bf16* __restrict__ Wt,
                      const float* __restrict__ bias, void* __restrict__ out,
                      int N) {
    const int lane = threadIdx.x & 31;
    const int wv   = threadIdx.x >> 5;   // 0..7
    const int hf   = lane >> 4;          // 0: lanes 0-15, 1: lanes 16-31
    const int l15  = lane & 15;
    const int m0   = blockIdx.x * 128 + (wv & 3) * 32;
    const int n0   = blockIdx.y * 128 + (wv >> 2) * 64;

    const __bf16* aP = X  + (size_t)(m0 + l15) * K + hf * 8;
    const __bf16* bP = Wt + (size_t)(n0 + l15) * K + hf * 8;

    v8f acc[2][4] = {};

    FragSet s0, s1;
    load_set<K>(s0, aP, bP, 0);
    load_set<K>(s1, aP, bP, 32);

    int k0 = 0;
#pragma unroll 1
    for (; k0 < K - 64; k0 += 64) {
        compute_set(s0, acc);
        load_set<K>(s0, aP, bP, k0 + 64);
        compute_set(s1, acc);
        load_set<K>(s1, aP, bP, k0 + 96);
        __builtin_prefetch(aP + k0 + 128, 0, 1);
    }
    compute_set(s0, acc);
    compute_set(s1, acc);

    // Epilogue: C/D layout — VGPR r, lane<16 -> (M=r, N=lane); lane>=16 -> (M=r+8)
#pragma unroll
    for (int i = 0; i < 2; ++i) {
#pragma unroll
        for (int j = 0; j < 4; ++j) {
            int n = n0 + j * 16 + l15;
            float bv = bias[n];
#pragma unroll
            for (int r = 0; r < 8; ++r) {
                int m = m0 + i * 16 + r + hf * 8;
                float v = acc[i][j][r] + bv;
                v = (v > 0.0f) ? v : SLOPE * v;
                if (LAST) {
                    ((float*)out)[(size_t)m * N + n] = 1.0f / (1.0f + __expf(-v));
                } else {
                    ((__bf16*)out)[(size_t)m * N + n] = (__bf16)v;
                }
            }
        }
    }
}

// ---------------------------------------------------------------------------
// Power iteration: one wave per 16x16 matrix (2048 matrices), 50 iterations.
// Lane l handles row (l&15); lanes 16-31 mirror lanes 0-15.
// ---------------------------------------------------------------------------
__global__ void power_iter_kernel(const float* __restrict__ pol, float* __restrict__ vec) {
    int wv   = threadIdx.x >> 5;
    int lane = threadIdx.x & 31;
    int mat  = blockIdx.x * 8 + wv;            // 0..2047
    int i    = lane & 15;
    const float* A = pol + (size_t)mat * 256;
    float a[16];
#pragma unroll
    for (int j = 0; j < 16; ++j) a[j] = A[i * 16 + j];
    float v = 1.0f;
    for (int it = 0; it < PI_ITERS; ++it) {
        float w = 0.0f;
#pragma unroll
        for (int j = 0; j < 16; ++j)
            w = fmaf(a[j], __shfl(v, j, 32), w);
        float s = w * w;
        s += __shfl_xor(s, 1, 32);
        s += __shfl_xor(s, 2, 32);
        s += __shfl_xor(s, 4, 32);
        s += __shfl_xor(s, 8, 32);
        v = w / sqrtf(s);
    }
    if (lane < 16) vec[(size_t)mat * 16 + i] = v;
}

// ---------------------------------------------------------------------------
// out[b][i] = sum_p vec[p,b,i] / vec[p,b,s(p)] * Ts[b, s(p), t(p)]
// ---------------------------------------------------------------------------
__global__ void combine_kernel(const float* __restrict__ vec, const float* __restrict__ Ts,
                               float* __restrict__ out) {
    int tid = threadIdx.x;       // 0..127
    int b = tid >> 4;
    int i = tid & 15;
    float s = 0.0f;
    for (int p = 0; p < PP; ++p) {
        int sid = p >> 4;
        const float* vp = vec + ((size_t)p * BB + b) * 16;
        s += vp[i] / vp[sid] * Ts[b * PP + p];
    }
    out[tid] = s;
}

// ---------------------------------------------------------------------------
extern "C" void kernel_launch(void* const* d_in, const int* in_sizes, int n_in,
                              void* d_out, int out_size, void* d_ws, size_t ws_size,
                              hipStream_t stream) {
    (void)in_sizes; (void)n_in; (void)out_size; (void)ws_size;
    const float* nodes = (const float*)d_in[0];
    // d_in[1] graphs_embeddings: unused by the reference
    const float* Ts = (const float*)d_in[2];
    const float* W1 = (const float*)d_in[3];
    const float* b1 = (const float*)d_in[4];
    const float* W2 = (const float*)d_in[5];
    const float* b2 = (const float*)d_in[6];
    const float* W3 = (const float*)d_in[7];
    const float* b3 = (const float*)d_in[8];
    const float* W4 = (const float*)d_in[9];
    const float* b4 = (const float*)d_in[10];
    float* out = (float*)d_out;

    char* ws = (char*)d_ws;
    auto carve = [&](size_t bytes) -> char* {
        char* p = ws;
        ws += (bytes + 255) & ~(size_t)255;
        return p;
    };
    __bf16* Xb   = (__bf16*)carve((size_t)MM * 256 * 2);        //  1 MB
    __bf16* W1t  = (__bf16*)carve((size_t)HH * 256 * 2);        //  2 MB
    __bf16* W4t  = (__bf16*)carve((size_t)PP * HH * 2);         //  2 MB
    __bf16* Wbig = (__bf16*)carve((size_t)HH * HH * 2);         // 32 MB (W2t, then W3t)
    __bf16* HA   = (__bf16*)carve((size_t)MM * HH * 2);         // 16 MB
    __bf16* HB   = (__bf16*)carve((size_t)MM * HH * 2);         // 16 MB
    float*  pol  = (float*)carve((size_t)MM * 256 * 4);         //  2 MB
    float*  vec  = (float*)carve((size_t)MM * 16 * 4);          // 128 KB

    // Stage inputs
    build_x_kernel<<<MM, 256, 0, stream>>>(nodes, Xb);
    transpose_cvt_kernel<<<dim3(256 / 32, HH / 32), 256, 0, stream>>>(W1, W1t, 256, HH);
    transpose_cvt_kernel<<<dim3(HH / 32, 256 / 32), 256, 0, stream>>>(W4, W4t, HH, 256);
    transpose_cvt_kernel<<<dim3(HH / 32, HH / 32), 256, 0, stream>>>(W2, Wbig, HH, HH);

    // MLP (grid: M/128 x N/128; each wave does 32x64)
    gemm_wmma_kernel<256,  false><<<dim3(MM / 128, HH / 128), 256, 0, stream>>>(Xb, W1t, b1, HA, HH);
    gemm_wmma_kernel<HH,   false><<<dim3(MM / 128, HH / 128), 256, 0, stream>>>(HA, Wbig, b2, HB, HH);
    transpose_cvt_kernel<<<dim3(HH / 32, HH / 32), 256, 0, stream>>>(W3, Wbig, HH, HH);
    gemm_wmma_kernel<HH,   false><<<dim3(MM / 128, HH / 128), 256, 0, stream>>>(HB, Wbig, b3, HA, HH);
    gemm_wmma_kernel<HH,   true ><<<dim3(MM / 128, 256 / 128), 256, 0, stream>>>(HA, W4t, b4, pol, 256);

    // Power iteration + final reduction
    power_iter_kernel<<<MM / 8, 256, 0, stream>>>(pol, vec);
    combine_kernel<<<1, 128, 0, stream>>>(vec, Ts, out);
}
